// SignedLogArray_9199819948570
// MI455X (gfx1250) — matrix operand order repaired and checked by reading, not compiled, not run on hardware
//
#include <hip/hip_runtime.h>
#include <math.h>

typedef __attribute__((ext_vector_type(16))) _Float16 v16h;
typedef __attribute__((ext_vector_type(8)))  _Float16 v8h;
typedef __attribute__((ext_vector_type(8)))  float    v8f;

#define MM 512
#define KK 512
#define NN 256

// ---------------------------------------------------------------------------
// Kernel 1: per-row max of a_log, then A'[m,k] = a_sign * exp(a_log - ra[m])
// stored as f16, row-major [M,K]. One block (256 threads) per row; each thread
// handles 2 columns; LDS tree reduction for the max.
// ---------------------------------------------------------------------------
__global__ void __launch_bounds__(256)
rowmax_scale_kernel(const float* __restrict__ a_log,
                    const float* __restrict__ a_sign,
                    _Float16* __restrict__ Ah,
                    float* __restrict__ ra) {
    __shared__ float red[256];
    const int m = blockIdx.x;
    const int t = threadIdx.x;

    const float la0 = a_log[m * KK + t];
    const float la1 = a_log[m * KK + t + 256];
    red[t] = fmaxf(la0, la1);
    __syncthreads();
    for (int s = 128; s > 0; s >>= 1) {
        if (t < s) red[t] = fmaxf(red[t], red[t + s]);
        __syncthreads();
    }
    const float rm = red[0];
    if (t == 0) ra[m] = rm;

    const float s0 = a_sign[m * KK + t];
    const float s1 = a_sign[m * KK + t + 256];
    Ah[m * KK + t]       = (_Float16)(s0 * __expf(la0 - rm));
    Ah[m * KK + t + 256] = (_Float16)(s1 * __expf(la1 - rm));
}

// ---------------------------------------------------------------------------
// Kernel 2: per-column max of b_log, then B' transposed:
//   Bt[n,k] = b_sign[k,n] * exp(b_log[k,n] - cb[n])  as f16, row-major [N,K].
// One thread per column n; the k-loop reads b_log/b_sign fully coalesced
// across threads. Transposed storage makes the GEMM B-operand loads
// contiguous 32-byte chunks per lane.
// ---------------------------------------------------------------------------
__global__ void __launch_bounds__(256)
colmax_scale_kernel(const float* __restrict__ b_log,
                    const float* __restrict__ b_sign,
                    _Float16* __restrict__ Bt,
                    float* __restrict__ cb) {
    const int n = threadIdx.x;            // gridDim.x == 1, blockDim.x == NN
    float mx = -__builtin_inff();
    for (int k = 0; k < KK; ++k) mx = fmaxf(mx, b_log[k * NN + n]);
    cb[n] = mx;
    for (int k = 0; k < KK; ++k) {
        const float lb = b_log[k * NN + n];   // L2-hot second pass
        const float sb = b_sign[k * NN + n];
        Bt[n * KK + k] = (_Float16)(sb * __expf(lb - mx));
    }
}

// ---------------------------------------------------------------------------
// Kernel 3: GEMM S = A' @ B'^T(Bt) via v_wmma_f32_16x16x32_f16, one 16x16
// output tile per wave, 8 waves per block. Epilogue:
//   out_log[m,n]  = ra[m] + cb[n] + log|S|
//   out_sign[m,n] = sign(S)
// written to d_out = [out_log (M*N) | out_sign (M*N)].
// ---------------------------------------------------------------------------
__global__ void __launch_bounds__(256)
wmma_gemm_kernel(const _Float16* __restrict__ Ah,
                 const _Float16* __restrict__ Bt,
                 const float* __restrict__ ra,
                 const float* __restrict__ cb,
                 float* __restrict__ out) {
    const int lane = threadIdx.x & 31;
    const int w    = threadIdx.x >> 5;
    const int mt   = blockIdx.x;                 // 0..31  (M/16)
    const int nt   = blockIdx.y * 8 + w;         // 0..15  (N/16)
    const int l15  = lane & 15;
    const int hi   = lane >> 4;                  // lane group (K-half select)

    const _Float16* ap = Ah + (mt * 16 + l15) * KK;   // my A row
    const _Float16* bp = Bt + (nt * 16 + l15) * KK;   // my B column (transposed)

    v8f c = {};
    #pragma unroll 4
    for (int k0 = 0; k0 < KK; k0 += 32) {
        // A 16x32 f16 layout: lanes 0-15 hold K [k0, k0+16),
        // lanes 16-31 hold K [k0+16, k0+32).
        const v16h a = *(const v16h*)(ap + k0 + 16 * hi);

        // B 32x16 f16 layout: lanes 0-15 hold K [k0, k0+16),
        // lanes 16-31 hold K [k0+16, k0+32), column = lane&15.
        const v16h b = *(const v16h*)(bp + k0 + 16 * hi);

        c = __builtin_amdgcn_wmma_f32_16x16x32_f16(
                /*neg_a=*/false, a, /*neg_b=*/false, b,
                /*c_mod=*/(short)0, c, /*reuse_a=*/false, /*reuse_b=*/false);
    }

    // C/D layout: VGPR r -> (row = r + 8*hi, col = lane&15) within tile.
    const int n   = nt * 16 + l15;
    const float cbn = cb[n];
    #pragma unroll
    for (int r = 0; r < 8; ++r) {
        const int m   = mt * 16 + r + 8 * hi;
        const float S = c[r];
        const float lg = ra[m] + cbn + __logf(fabsf(S));
        const float sg = (S > 0.0f) ? 1.0f : ((S < 0.0f) ? -1.0f : 0.0f);
        out[m * NN + n]           = lg;   // out_log
        out[MM * NN + m * NN + n] = sg;   // out_sign
    }
}

extern "C" void kernel_launch(void* const* d_in, const int* in_sizes, int n_in,
                              void* d_out, int out_size, void* d_ws, size_t ws_size,
                              hipStream_t stream) {
    const float* a_log  = (const float*)d_in[0];
    const float* a_sign = (const float*)d_in[1];
    const float* b_log  = (const float*)d_in[2];
    const float* b_sign = (const float*)d_in[3];
    float* out = (float*)d_out;

    // Workspace layout (bytes): Ah f16 [M*K] | Bt f16 [N*K] | ra f32[M] | cb f32[N]
    char* ws = (char*)d_ws;
    _Float16* Ah = (_Float16*)ws;
    _Float16* Bt = (_Float16*)(ws + (size_t)MM * KK * sizeof(_Float16));
    float*    ra = (float*)(ws + (size_t)MM * KK * sizeof(_Float16)
                               + (size_t)NN * KK * sizeof(_Float16));
    float*    cb = ra + MM;

    rowmax_scale_kernel<<<MM, 256, 0, stream>>>(a_log, a_sign, Ah, ra);
    colmax_scale_kernel<<<1, NN, 0, stream>>>(b_log, b_sign, Bt, cb);
    wmma_gemm_kernel<<<dim3(MM / 16, NN / (16 * 8)), 256, 0, stream>>>(Ah, Bt, ra, cb, out);
}
